// Attention_49813030699234
// MI455X (gfx1250) — matrix operand order repaired and checked by reading, not compiled, not run on hardware
//
#include <hip/hip_runtime.h>
#include <hip/hip_fp16.h>

#define BATCH 4
#define NTOK  2304
#define CDIM  256
#define NHEAD 8
#define HD    32
#define HS    48
#define MROWS (BATCH * NTOK)       /* 9216 */
#define SCALE 0.0625f              /* 256^-0.5 */
#define L2E   1.4426950408889634f
#define PSTR  40                   /* LDS P row stride (halves); 80B = 5*16B keeps 16B align */

typedef __attribute__((ext_vector_type(16))) _Float16 v16h;
typedef __attribute__((ext_vector_type(8)))  _Float16 v8h;
typedef __attribute__((ext_vector_type(8)))  float    v8f;

// ---- WMMA fragment loaders (contiguous k order, same for A and B) ----
// A-matrix 16x32 f16: lane (g = lane>>4, m = lane&15) holds row m,
// K = [16g, 16g+16) contiguous.
__device__ __forceinline__ v16h ld_a_frag(const _Float16* row, int g) {
  v8h lo = *(const v8h*)(row + 16 * g);
  v8h hi = *(const v8h*)(row + 16 * g + 8);
  v16h r;
#pragma unroll
  for (int i = 0; i < 8; ++i) { r[i] = lo[i]; r[i + 8] = hi[i]; }
  return r;
}
// B-matrix 32x16 f16: lane holds column n = lane&15, K = [16g, 16g+16) contiguous.
__device__ __forceinline__ v16h ld_b_frag(const _Float16* col, int g) {
  return *(const v16h*)(col + 16 * g);
}
__device__ __forceinline__ v8f wmma16(v16h a, v16h b, v8f c) {
  return __builtin_amdgcn_wmma_f32_16x16x32_f16(false, a, false, b, (short)0, c,
                                                false, false);
}

// ---- Kernel 1: fused depthwise 3x3 conv for q/k/v streams, f32 -> f16 ----
// grid = (NTOK*CDIM/256, BATCH): no div/mod by NTOK needed.
__global__ void dwconv_qkv(const float* __restrict__ x,
                           const float* __restrict__ wq,
                           const float* __restrict__ wk,
                           const float* __restrict__ wv,
                           _Float16* __restrict__ cq,
                           _Float16* __restrict__ ck,
                           _Float16* __restrict__ cv) {
  int lid = blockIdx.x * 256 + threadIdx.x;      // enumerates (t*C + c) within batch
  int b   = blockIdx.y;
  int c   = lid & (CDIM - 1);
  int t   = lid >> 8;                             // < 2304
  int y = t / HS, xx = t % HS;                    // /48: cheap, small range
  float aq = 0.f, ak = 0.f, av = 0.f;
#pragma unroll
  for (int dy = -1; dy <= 1; ++dy) {
#pragma unroll
    for (int dx = -1; dx <= 1; ++dx) {
      int ny = y + dy, nx = xx + dx;
      if (ny >= 0 && ny < HS && nx >= 0 && nx < HS) {
        float v = x[((size_t)(b * NTOK + ny * HS + nx)) * CDIM + c];
        int wi = c * 9 + (dy + 1) * 3 + (dx + 1);
        aq = fmaf(v, wq[wi], aq);
        ak = fmaf(v, wk[wi], ak);
        av = fmaf(v, wv[wi], av);
      }
    }
  }
  size_t o = (size_t)(b * NTOK + t) * CDIM + c;
  cq[o] = (_Float16)aq;
  ck[o] = (_Float16)ak;
  cv[o] = (_Float16)av;
}

// ---- Kernel 2: convert the four 256x256 weight matrices to f16 ----
__global__ void cvt_weights(const float* __restrict__ Wq, const float* __restrict__ Wk,
                            const float* __restrict__ Wv, const float* __restrict__ Wp,
                            _Float16* __restrict__ hW) {
  int i = blockIdx.x * 256 + threadIdx.x;        // 4 * 65536 elements
  int m = i >> 16;
  int o = i & 0xFFFF;
  const float* src = (m == 0) ? Wq : (m == 1) ? Wk : (m == 2) ? Wv : Wp;
  hW[i] = (_Float16)src[o];
}

// ---- Kernel 3: projection GEMM  y = tokens @ W^T, scatter to head layout ----
// grid = (576, BATCH), 4 waves/block, one 16x16 tile per wave.
// vmode 0: dst[b][h][t][d]   (Q, K)
// vmode 1: dst[b][h][d][t]   (V transposed: O-GEMM B-fragments contiguous in keys)
__global__ void proj_qkv(const _Float16* __restrict__ src,
                         const _Float16* __restrict__ W,
                         _Float16* __restrict__ dst, int vmode) {
  int wave = threadIdx.x >> 5, lane = threadIdx.x & 31;
  int g = lane >> 4, mr = lane & 15;
  int b_ = blockIdx.y;
  int tile = blockIdx.x * 4 + wave;              // 144*16 = 2304 tiles per batch
  int tm = tile >> 4, tn = tile & 15;
  int t0 = tm * 16;                              // token tile base within batch
  const _Float16* ap = src + (size_t)(b_ * NTOK + t0 + mr) * CDIM;
  const _Float16* bq = W + (size_t)(tn * 16 + mr) * CDIM;   // W[out][in], col n = out
  v8f acc = {};
#pragma unroll
  for (int kb = 0; kb < CDIM; kb += 32) {
    v16h a = ld_a_frag(ap + kb, g);
    v16h b = ld_b_frag(bq + kb, g);
    acc = wmma16(a, b, acc);
  }
  int col = tn * 16 + mr;
  int h = col >> 5, d = col & 31;
  size_t bh = (size_t)(b_ * NHEAD + h);
#pragma unroll
  for (int r = 0; r < 8; ++r) {
    int t = t0 + r + 8 * g;
    _Float16 val = (_Float16)acc[r];
    if (vmode == 0)
      dst[(bh * NTOK + t) * HD + d] = val;
    else
      dst[(bh * HD + d) * NTOK + t] = val;
  }
}

// ---- Kernel 4: flash attention; grid = (36, 32 bh-slices), 4 waves/block ----
__global__ void attn_flash(const _Float16* __restrict__ q,
                           const _Float16* __restrict__ k,
                           const _Float16* __restrict__ vT,
                           _Float16* __restrict__ ao) {
  __shared__ alignas(32) _Float16 ldsP[4][16 * PSTR];
  int wave = threadIdx.x >> 5, lane = threadIdx.x & 31;
  int g = lane >> 4, mr = lane & 15;
  int bh = blockIdx.y;                           // 32 (b,h) slices
  int q0 = blockIdx.x * 64 + wave * 16;
  int b_ = bh >> 3, h = bh & 7;

  const _Float16* kbase = k + (size_t)bh * NTOK * HD;
  const _Float16* vbase = vT + (size_t)bh * HD * NTOK;

  v16h qa = ld_a_frag(q + ((size_t)bh * NTOK + q0 + mr) * HD, g);

  v8f o1 = {}, o2 = {};
  float m_run[8], l_run[8];
#pragma unroll
  for (int r = 0; r < 8; ++r) { m_run[r] = -INFINITY; l_run[r] = 0.f; }

  _Float16* lp = ldsP[wave];
  for (int k0 = 0; k0 < NTOK; k0 += 32) {
    // S = Q K^T for 32 keys (two 16-key B fragments, full K = d = 32)
    v16h kb1 = ld_b_frag(kbase + (size_t)(k0 + mr) * HD, g);
    v16h kb2 = ld_b_frag(kbase + (size_t)(k0 + 16 + mr) * HD, g);
    // pull next chunk's K/V lines toward the WGP while softmax VALU runs
    // (global_prefetch_b8; WMMA-f16 co-executes with VALU so this latency hides)
    if (k0 + 32 < NTOK) {
      __builtin_prefetch(kbase + (size_t)(k0 + 32 + mr) * HD, 0, 3);
      __builtin_prefetch(vbase + (size_t)mr * NTOK + k0 + 32, 0, 3);
    }
    v8f s1 = {}, s2 = {};
    s1 = wmma16(qa, kb1, s1);
    s2 = wmma16(qa, kb2, s2);

    // online softmax: C-fragment row = r + 8g, col = lane&15
#pragma unroll
    for (int r = 0; r < 8; ++r) {
      float a1 = s1[r] * SCALE;
      float a2 = s2[r] * SCALE;
      float xm = fmaxf(a1, a2);
      xm = fmaxf(xm, __shfl_xor(xm, 1, 32));
      xm = fmaxf(xm, __shfl_xor(xm, 2, 32));
      xm = fmaxf(xm, __shfl_xor(xm, 4, 32));
      xm = fmaxf(xm, __shfl_xor(xm, 8, 32));
      float mo = m_run[r];
      float mn = fmaxf(mo, xm);
      float corr = __builtin_amdgcn_exp2f((mo - mn) * L2E);
      float p1 = __builtin_amdgcn_exp2f((a1 - mn) * L2E);
      float p2 = __builtin_amdgcn_exp2f((a2 - mn) * L2E);
      float rs = p1 + p2;
      rs += __shfl_xor(rs, 1, 32);
      rs += __shfl_xor(rs, 2, 32);
      rs += __shfl_xor(rs, 4, 32);
      rs += __shfl_xor(rs, 8, 32);
      l_run[r] = l_run[r] * corr + rs;
      m_run[r] = mn;
      o1[r] *= corr;
      o2[r] *= corr;
      s1[r] = p1;
      s2[r] = p2;
    }

    // transpose P (C layout) -> A-fragment layout via LDS
#pragma unroll
    for (int r = 0; r < 8; ++r) {
      int row = r + 8 * g;
      lp[row * PSTR + mr]      = (_Float16)s1[r];
      lp[row * PSTR + 16 + mr] = (_Float16)s2[r];
    }
    __syncthreads();
    v16h pa = ld_a_frag(lp + mr * PSTR, g);
    v16h vb1 = ld_b_frag(vbase + (size_t)mr * NTOK + k0, g);
    v16h vb2 = ld_b_frag(vbase + (size_t)(16 + mr) * NTOK + k0, g);
    o1 = wmma16(pa, vb1, o1);
    o2 = wmma16(pa, vb2, o2);
    __syncthreads();
  }

  // normalize and scatter back to token-major [B*N, C] f16 for the Wp GEMM
#pragma unroll
  for (int r = 0; r < 8; ++r) {
    float inv = 1.0f / l_run[r];
    int t = q0 + r + 8 * g;
    size_t base = ((size_t)(b_ * NTOK + t)) * CDIM + h * HD;
    ao[base + mr]      = (_Float16)(o1[r] * inv);
    ao[base + 16 + mr] = (_Float16)(o2[r] * inv);
  }
}

// ---- Kernel 5: final projection y = ao @ Wp^T + bp, f32 output ----
// grid = (576, BATCH), 4 waves/block.
__global__ void proj_final(const _Float16* __restrict__ src,
                           const _Float16* __restrict__ W,
                           const float* __restrict__ bias,
                           float* __restrict__ out) {
  int wave = threadIdx.x >> 5, lane = threadIdx.x & 31;
  int g = lane >> 4, mr = lane & 15;
  int b_ = blockIdx.y;
  int tile = blockIdx.x * 4 + wave;
  int tm = tile >> 4, tn = tile & 15;
  int t0 = tm * 16;
  const _Float16* ap = src + (size_t)(b_ * NTOK + t0 + mr) * CDIM;
  const _Float16* bq = W + (size_t)(tn * 16 + mr) * CDIM;
  v8f acc = {};
#pragma unroll
  for (int kb = 0; kb < CDIM; kb += 32) {
    v16h a = ld_a_frag(ap + kb, g);
    v16h b = ld_b_frag(bq + kb, g);
    acc = wmma16(a, b, acc);
  }
  int col = tn * 16 + mr;
  float bv = bias[col];
#pragma unroll
  for (int r = 0; r < 8; ++r) {
    size_t grow = (size_t)(b_ * NTOK + t0 + r + 8 * g);
    out[grow * CDIM + col] = acc[r] + bv;
  }
}

extern "C" void kernel_launch(void* const* d_in, const int* in_sizes, int n_in,
                              void* d_out, int out_size, void* d_ws, size_t ws_size,
                              hipStream_t stream) {
  const float* x   = (const float*)d_in[0];
  const float* wqc = (const float*)d_in[1];
  const float* wkc = (const float*)d_in[2];
  const float* wvc = (const float*)d_in[3];
  const float* Wq  = (const float*)d_in[4];
  const float* Wk  = (const float*)d_in[5];
  const float* Wv  = (const float*)d_in[6];
  const float* Wp  = (const float*)d_in[7];
  const float* bp  = (const float*)d_in[8];
  float* out = (float*)d_out;

  // workspace carve-up (all f16): 7 token-sized buffers + weights = ~33.6 MB
  const size_t SZT = (size_t)MROWS * CDIM * sizeof(_Float16);   // 4.72 MB
  const size_t WSZ = (size_t)4 * CDIM * CDIM * sizeof(_Float16);
  char* w = (char*)d_ws;
  _Float16* cq  = (_Float16*)(w);
  _Float16* ck  = (_Float16*)(w + SZT);
  _Float16* cv  = (_Float16*)(w + 2 * SZT);
  _Float16* hW  = (_Float16*)(w + 3 * SZT);
  _Float16* qb  = (_Float16*)(w + 3 * SZT + WSZ);
  _Float16* kb  = (_Float16*)(w + 4 * SZT + WSZ);
  _Float16* vTb = (_Float16*)(w + 5 * SZT + WSZ);
  _Float16* ao  = (_Float16*)(w + 6 * SZT + WSZ);
  _Float16* hWq = hW;
  _Float16* hWk = hW + CDIM * CDIM;
  _Float16* hWv = hW + 2 * CDIM * CDIM;
  _Float16* hWp = hW + 3 * CDIM * CDIM;

  dwconv_qkv<<<dim3(NTOK * CDIM / 256, BATCH), 256, 0, stream>>>(x, wqc, wkc, wvc,
                                                                 cq, ck, cv);
  cvt_weights<<<(4 * CDIM * CDIM) / 256, 256, 0, stream>>>(Wq, Wk, Wv, Wp, hW);

  // 2304 tiles per batch, 4 waves/block -> (576, BATCH) blocks per GEMM
  proj_qkv<<<dim3(576, BATCH), 128, 0, stream>>>(cq, hWq, qb, 0);
  proj_qkv<<<dim3(576, BATCH), 128, 0, stream>>>(ck, hWk, kb, 0);
  proj_qkv<<<dim3(576, BATCH), 128, 0, stream>>>(cv, hWv, vTb, 1);

  // 36 query blocks x 32 (b,h) slices
  attn_flash<<<dim3(36, BATCH * NHEAD), 128, 0, stream>>>(qb, kb, vTb, ao);

  proj_final<<<dim3(576, BATCH), 128, 0, stream>>>(ao, hWp, bp, out);
}
